// GAT_WAVE_70420283785461
// MI455X (gfx1250) — compile-verified, matrix operand order checked by reading, not run on hardware
//
#include <hip/hip_runtime.h>
#include <math.h>

typedef __attribute__((ext_vector_type(16))) _Float16 v16h;
typedef __attribute__((ext_vector_type(8)))  float    v8f;

#define NN   2048   // nodes
#define BB   8      // batch
#define CC   64     // channels
#define HH   2      // heads
#define HC   128    // H*C
#define NEG_SLOPE 0.2f
#define LN_EPS 1e-5f

// ---------------------------------------------------------------------------
// Generic WMMA GEMM, one wave computes a 16x32 output tile (two accumulators,
// A fragment reused across both N-subtiles):
//   C[M,Nn] (+)= A' * B'  with
//   A'[m,k] = A[remap(m)*K + k],  remap(m) = (m/rpg_out)*rpg_in + m%rpg_out + row_off
//   B'[k,n] = Bw[b_off + k*b_sk + n*b_sn]
// M must be a multiple of 16; Nn is masked. f32->f16 in-register conversion,
// f32 accumulation via v_wmma_f32_16x16x32_f16.
// ---------------------------------------------------------------------------
__global__ __launch_bounds__(32)
void wmma_gemm_kernel(const float* __restrict__ A,
                      const float* __restrict__ Bw,
                      const float* __restrict__ bias,
                      float* __restrict__ C,
                      int M, int K, int Nn,
                      int rpg_out, int rpg_in, int row_off,
                      int b_sk, int b_sn, int b_off,
                      int accumulate, int relu_act)
{
    const int tile_m = blockIdx.x;
    const int tile_n = blockIdx.y;          // 32 columns per tile
    const int lane   = threadIdx.x;
    const int half   = lane >> 4;           // 0: K 0-7/16-23 ; 1: K 8-15/24-31
    const int l15    = lane & 15;

    const int mrow = tile_m * 16 + l15;     // A-fragment row of this lane
    const int arow = (mrow / rpg_out) * rpg_in + (mrow % rpg_out) + row_off;
    const float* __restrict__ Arow = A + (long)arow * K;

    const int  ncol0 = tile_n * 32 + l15;
    const int  ncol1 = ncol0 + 16;
    const bool nok0  = ncol0 < Nn;
    const bool nok1  = ncol1 < Nn;

    v8f c0, c1;
#pragma unroll
    for (int r = 0; r < 8; ++r) { c0[r] = 0.0f; c1[r] = 0.0f; }
    if (accumulate) {
#pragma unroll
        for (int r = 0; r < 8; ++r) {
            int m = tile_m * 16 + r + 8 * half;
            if (nok0) c0[r] = C[(long)m * Nn + ncol0];
            if (nok1) c1[r] = C[(long)m * Nn + ncol1];
        }
    }

    for (int k0 = 0; k0 < K; k0 += 32) {
        const int kb1 = k0 + half * 8;      // aligned to 8 floats (32B)
        const int kb2 = k0 + 16 + half * 8;

        // A fragment: 4x float4 vector loads, reused by both wmma's
        const float4 a0 = *(const float4*)&Arow[kb1];
        const float4 a1 = *(const float4*)&Arow[kb1 + 4];
        const float4 a2 = *(const float4*)&Arow[kb2];
        const float4 a3 = *(const float4*)&Arow[kb2 + 4];
        v16h a;
        a[0]=(_Float16)a0.x; a[1]=(_Float16)a0.y; a[2]=(_Float16)a0.z; a[3]=(_Float16)a0.w;
        a[4]=(_Float16)a1.x; a[5]=(_Float16)a1.y; a[6]=(_Float16)a1.z; a[7]=(_Float16)a1.w;
        a[8]=(_Float16)a2.x; a[9]=(_Float16)a2.y; a[10]=(_Float16)a2.z; a[11]=(_Float16)a2.w;
        a[12]=(_Float16)a3.x; a[13]=(_Float16)a3.y; a[14]=(_Float16)a3.z; a[15]=(_Float16)a3.w;

        v16h b0, b1;
        if (nok0) {
            const float* __restrict__ Bp = Bw + b_off + (long)ncol0 * b_sn;
#pragma unroll
            for (int e = 0; e < 8; ++e) {
                b0[e]     = (_Float16)Bp[(long)(kb1 + e) * b_sk];
                b0[e + 8] = (_Float16)Bp[(long)(kb2 + e) * b_sk];
            }
        } else {
#pragma unroll
            for (int e = 0; e < 16; ++e) b0[e] = (_Float16)0.0f;
        }
        if (nok1) {
            const float* __restrict__ Bp = Bw + b_off + (long)ncol1 * b_sn;
#pragma unroll
            for (int e = 0; e < 8; ++e) {
                b1[e]     = (_Float16)Bp[(long)(kb1 + e) * b_sk];
                b1[e + 8] = (_Float16)Bp[(long)(kb2 + e) * b_sk];
            }
        } else {
#pragma unroll
            for (int e = 0; e < 16; ++e) b1[e] = (_Float16)0.0f;
        }

        c0 = __builtin_amdgcn_wmma_f32_16x16x32_f16(false, a, false, b0,
                                                    (short)0, c0, false, false);
        c1 = __builtin_amdgcn_wmma_f32_16x16x32_f16(false, a, false, b1,
                                                    (short)0, c1, false, false);
    }

    const float bv0 = (bias && nok0) ? bias[ncol0] : 0.0f;
    const float bv1 = (bias && nok1) ? bias[ncol1] : 0.0f;
#pragma unroll
    for (int r = 0; r < 8; ++r) {
        int m = tile_m * 16 + r + 8 * half;
        if (nok0) {
            float v = c0[r] + bv0;
            if (relu_act) v = fmaxf(v, 0.0f);
            C[(long)m * Nn + ncol0] = v;
        }
        if (nok1) {
            float v = c1[r] + bv1;
            if (relu_act) v = fmaxf(v, 0.0f);
            C[(long)m * Nn + ncol1] = v;
        }
    }
}

// ---------------------------------------------------------------------------
// el/er: per (b,n,t,h) 64-length dot products with attn vectors (float4 path)
// feat layout: [B,N,t,H*C]; el/er layout: [B,N,t,H]
// ---------------------------------------------------------------------------
__global__ __launch_bounds__(256)
void attn_proj_kernel(const float* __restrict__ feat,
                      const float* __restrict__ al,
                      const float* __restrict__ ar,
                      float* __restrict__ el, float* __restrict__ er,
                      int rows)
{
    int gid = blockIdx.x * 256 + threadIdx.x;
    if (gid >= rows * HH) return;
    int row = gid / HH, h = gid % HH;
    const float4* __restrict__ f4 = (const float4*)(feat + (long)row * HC + h * CC);
    const float4* __restrict__ l4 = (const float4*)(al + h * CC);
    const float4* __restrict__ r4 = (const float4*)(ar + h * CC);
    float sl = 0.0f, sr = 0.0f;
#pragma unroll 4
    for (int ci = 0; ci < CC / 4; ++ci) {
        float4 v = f4[ci], la = l4[ci], ra = r4[ci];
        sl += v.x * la.x + v.y * la.y + v.z * la.z + v.w * la.w;
        sr += v.x * ra.x + v.y * ra.y + v.z * ra.z + v.w * ra.w;
    }
    el[gid] = sl;
    er[gid] = sr;
}

__global__ __launch_bounds__(256)
void fill4_kernel(float4* __restrict__ p, float v, int n4)
{
    int gid = blockIdx.x * 256 + threadIdx.x;
    if (gid < n4) p[gid] = make_float4(v, v, v, v);
}

__device__ inline void atomicMaxF(float* addr, float val)
{
    int cur = __float_as_int(*addr);
    while (__int_as_float(cur) < val) {
        int assumed = cur;
        cur = atomicCAS((int*)addr, assumed, __float_as_int(val));
        if (cur == assumed) break;
    }
}

// Pass 1: e = leaky_relu(el[src]+er[dst]); store; segment max into mbuf[dst]
__global__ __launch_bounds__(256)
void edge_logit_max_kernel(const int* __restrict__ src, const int* __restrict__ dst,
                           const float* __restrict__ el, const float* __restrict__ er,
                           float* __restrict__ evals, float* __restrict__ mbuf,
                           int E, int BTH, int tH)
{
    int gid = blockIdx.x * 256 + threadIdx.x;
    if (gid >= E * BTH) return;
    int edge = gid / BTH, bth = gid % BTH;
    int b = bth / tH, r = bth % tH;
    int s = src[edge], d = dst[edge];
    float e = el[(long)(b * NN + s) * tH + r] + er[(long)(b * NN + d) * tH + r];
    e = (e > 0.0f) ? e : NEG_SLOPE * e;
    evals[gid] = e;
    atomicMaxF(&mbuf[(long)d * BTH + bth], e);
}

// Pass 2: a = exp(e - m[dst]); store; segment sum into ssum[dst]
__global__ __launch_bounds__(256)
void edge_exp_sum_kernel(const int* __restrict__ dst,
                         float* __restrict__ evals,
                         const float* __restrict__ mbuf, float* __restrict__ ssum,
                         int E, int BTH)
{
    int gid = blockIdx.x * 256 + threadIdx.x;
    if (gid >= E * BTH) return;
    int edge = gid / BTH, bth = gid % BTH;
    int d = dst[edge];
    float a = __expf(evals[gid] - mbuf[(long)d * BTH + bth]);
    evals[gid] = a;
    atomicAdd(&ssum[(long)d * BTH + bth], a);
}

// Pass 3: out[dst] += feat[src] * (a/ssum[dst]); 4 channels per thread
__global__ __launch_bounds__(256)
void edge_aggregate_kernel(const int* __restrict__ src, const int* __restrict__ dst,
                           const float* __restrict__ feat,
                           const float* __restrict__ evals, const float* __restrict__ ssum,
                           float* __restrict__ outb,
                           int E, int BTH, int tH, int tdim)
{
    int gid = blockIdx.x * 256 + threadIdx.x;
    int total = E * BTH * (CC / 4);
    if (gid >= total) return;
    int c4 = (gid & 15) * 4;
    int t  = gid >> 4;                 // (edge, bth)
    int edge = t / BTH, bth = t % BTH;
    int b = bth / tH, r = bth % tH;
    int tt = r >> 1, h = r & 1;
    int s = src[edge], d = dst[edge];
    float alpha = evals[t] / ssum[(long)d * BTH + bth];
    long fidx = ((long)(b * NN + s) * tdim + tt) * HC + h * CC + c4;
    long oidx = ((long)(b * NN + d) * tdim + tt) * HC + h * CC + c4;
    float4 f = *(const float4*)&feat[fidx];
    atomicAdd(&outb[oidx + 0], f.x * alpha);
    atomicAdd(&outb[oidx + 1], f.y * alpha);
    atomicAdd(&outb[oidx + 2], f.z * alpha);
    atomicAdd(&outb[oidx + 3], f.w * alpha);
}

// out = relu(out + bias[HC]) over [.., HC] rows, float4-wide
__global__ __launch_bounds__(256)
void bias_relu_kernel(float4* __restrict__ p, const float* __restrict__ bias, int total4)
{
    int gid = blockIdx.x * 256 + threadIdx.x;
    if (gid >= total4) return;
    float4 v = p[gid];
    const float4 bv = *(const float4*)&bias[(gid * 4) % HC];
    v.x = fmaxf(v.x + bv.x, 0.0f);
    v.y = fmaxf(v.y + bv.y, 0.0f);
    v.z = fmaxf(v.z + bv.z, 0.0f);
    v.w = fmaxf(v.w + bv.w, 0.0f);
    p[gid] = v;
}

// LayerNorm over (N,C) jointly for each (b,t). x,y layout: [B,N,Tdim,C]
__global__ __launch_bounds__(256)
void ln_nc_kernel(const float* __restrict__ x, float* __restrict__ y,
                  const float* __restrict__ w, const float* __restrict__ bb,
                  int Tdim)
{
    __shared__ float s1[256], s2[256];
    const int bt = blockIdx.x;
    const int b = bt / Tdim, t = bt % Tdim;
    const int NC = NN * CC;
    const int tid = threadIdx.x;
    float sum = 0.0f, sq = 0.0f;
    for (int i = tid * 4; i < NC; i += 1024) {
        int n = i >> 6, c = i & 63;             // c multiple of 4, within-row
        float4 v = *(const float4*)&x[((long)(b * NN + n) * Tdim + t) * CC + c];
        sum += v.x + v.y + v.z + v.w;
        sq  += v.x * v.x + v.y * v.y + v.z * v.z + v.w * v.w;
    }
    s1[tid] = sum; s2[tid] = sq;
    __syncthreads();
    for (int st = 128; st > 0; st >>= 1) {
        if (tid < st) { s1[tid] += s1[tid + st]; s2[tid] += s2[tid + st]; }
        __syncthreads();
    }
    const float mean = s1[0] / (float)NC;
    const float var  = s2[0] / (float)NC - mean * mean;
    const float rs   = rsqrtf(var + LN_EPS);
    for (int i = tid * 4; i < NC; i += 1024) {
        int n = i >> 6, c = i & 63;
        long idx = ((long)(b * NN + n) * Tdim + t) * CC + c;
        float4 v = *(const float4*)&x[idx];
        float4 ww = *(const float4*)&w[i];
        float4 bv = *(const float4*)&bb[i];
        v.x = (v.x - mean) * rs * ww.x + bv.x;
        v.y = (v.y - mean) * rs * ww.y + bv.y;
        v.z = (v.z - mean) * rs * ww.z + bv.z;
        v.w = (v.w - mean) * rs * ww.w + bv.w;
        *(float4*)&y[idx] = v;
    }
}

// Torch-faithful reshape: xt2 logically [B,C,N,12] -> [B, N, C*12];
// our xt2 is stored [B,N,12,C].
__global__ __launch_bounds__(256)
void repack_fc_kernel(const float* __restrict__ xt2, float* __restrict__ out, int total)
{
    int gid = blockIdx.x * 256 + threadIdx.x;
    if (gid >= total) return;
    int per_b = NN * 768;
    int b = gid / per_b;
    int idx = gid % per_b;                 // linear over [C, N, 12]
    int cprime = idx / (NN * 12);
    int rem = idx % (NN * 12);
    int nprime = rem / 12, jprime = rem % 12;
    out[gid] = xt2[((long)(b * NN + nprime) * 12 + jprime) * CC + cprime];
}

// ---------------------------------------------------------------------------
extern "C" void kernel_launch(void* const* d_in, const int* in_sizes, int n_in,
                              void* d_out, int out_size, void* d_ws, size_t ws_size,
                              hipStream_t stream)
{
    (void)n_in; (void)out_size; (void)ws_size;
    const float* x    = (const float*)d_in[0];
    const int*   src  = (const int*)d_in[1];
    const int*   dst  = (const int*)d_in[2];
    const float* W1   = (const float*)d_in[3];
    const float* al1  = (const float*)d_in[4];
    const float* ar1  = (const float*)d_in[5];
    const float* b1   = (const float*)d_in[6];
    const float* W2   = (const float*)d_in[7];
    const float* al2  = (const float*)d_in[8];
    const float* ar2  = (const float*)d_in[9];
    const float* b2   = (const float*)d_in[10];
    const float* ln1w = (const float*)d_in[11];
    const float* ln1b = (const float*)d_in[12];
    const float* ln2w = (const float*)d_in[13];
    const float* ln2b = (const float*)d_in[14];
    const float* tc1w = (const float*)d_in[15];
    const float* tc1b = (const float*)d_in[16];
    const float* ln3w = (const float*)d_in[17];
    const float* ln3b = (const float*)d_in[18];
    const float* tc2w = (const float*)d_in[19];
    const float* tc2b = (const float*)d_in[20];
    const float* fcw  = (const float*)d_in[21];
    const float* fcb  = (const float*)d_in[22];
    const int E = in_sizes[1];

    // workspace carve-up (floats)
    float* ws = (float*)d_ws;
    const size_t BIG = (size_t)BB * NN * 16 * CC;   // 16,777,216
    float* bufA  = ws;                 // [B,N,16,C] max
    float* bufB  = bufA + BIG;         // [B,N,16,C] max
    float* featb = bufB + BIG;         // [B,N,8,HC] max = BIG
    float* evals = featb + BIG;        // E * 128 max
    float* mbuf  = evals + (size_t)E * 128;
    float* ssum  = mbuf  + (size_t)NN * 128;
    float* elb   = ssum  + (size_t)NN * 128;
    float* erb   = elb   + (size_t)NN * 128;        // [B,N,8,H] max = 262144 <= NN*128

    const int THREADS = 256;
    #define GRID1(n) dim3(((n) + THREADS - 1) / THREADS)

    // ===================== GAT layer 1 (tdim = 4) =====================
    {
        const int tdim = 4, tH = tdim * HH, BTH = BB * tH;
        const int M = BB * NN * tdim;
        // feat = x @ W1  : [M,64] @ [64,128]
        wmma_gemm_kernel<<<dim3(M / 16, HC / 32), 32, 0, stream>>>(
            x, W1, nullptr, featb, M, CC, HC, 1, 1, 0, HC, 1, 0, 0, 0);
        attn_proj_kernel<<<GRID1(M * HH), THREADS, 0, stream>>>(featb, al1, ar1, elb, erb, M);

        const int aggN = BB * NN * tdim * HC;   // 8,388,608
        fill4_kernel<<<GRID1(NN * BTH / 4), THREADS, 0, stream>>>((float4*)mbuf, -1e30f, NN * BTH / 4);
        fill4_kernel<<<GRID1(NN * BTH / 4), THREADS, 0, stream>>>((float4*)ssum, 0.0f, NN * BTH / 4);
        fill4_kernel<<<GRID1(aggN / 4), THREADS, 0, stream>>>((float4*)bufA, 0.0f, aggN / 4);

        edge_logit_max_kernel<<<GRID1(E * BTH), THREADS, 0, stream>>>(
            src, dst, elb, erb, evals, mbuf, E, BTH, tH);
        edge_exp_sum_kernel<<<GRID1(E * BTH), THREADS, 0, stream>>>(
            dst, evals, mbuf, ssum, E, BTH);
        edge_aggregate_kernel<<<GRID1(E * BTH * 16), THREADS, 0, stream>>>(
            src, dst, featb, evals, ssum, bufA, E, BTH, tH, tdim);
        bias_relu_kernel<<<GRID1(aggN / 4), THREADS, 0, stream>>>((float4*)bufA, b1, aggN / 4);
        // LN1 over (N,C) per (b, t*H=8) : bufA -> bufB  [B,N,8,C]
        ln_nc_kernel<<<BB * 8, THREADS, 0, stream>>>(bufA, bufB, ln1w, ln1b, 8);
    }

    // ===================== GAT layer 2 (tdim = 8) =====================
    {
        const int tdim = 8, tH = tdim * HH, BTH = BB * tH;
        const int M = BB * NN * tdim;
        wmma_gemm_kernel<<<dim3(M / 16, HC / 32), 32, 0, stream>>>(
            bufB, W2, nullptr, featb, M, CC, HC, 1, 1, 0, HC, 1, 0, 0, 0);
        attn_proj_kernel<<<GRID1(M * HH), THREADS, 0, stream>>>(featb, al2, ar2, elb, erb, M);

        const int aggN = BB * NN * tdim * HC;   // 16,777,216
        fill4_kernel<<<GRID1(NN * BTH / 4), THREADS, 0, stream>>>((float4*)mbuf, -1e30f, NN * BTH / 4);
        fill4_kernel<<<GRID1(NN * BTH / 4), THREADS, 0, stream>>>((float4*)ssum, 0.0f, NN * BTH / 4);
        fill4_kernel<<<GRID1(aggN / 4), THREADS, 0, stream>>>((float4*)bufA, 0.0f, aggN / 4);

        edge_logit_max_kernel<<<GRID1(E * BTH), THREADS, 0, stream>>>(
            src, dst, elb, erb, evals, mbuf, E, BTH, tH);
        edge_exp_sum_kernel<<<GRID1(E * BTH), THREADS, 0, stream>>>(
            dst, evals, mbuf, ssum, E, BTH);
        edge_aggregate_kernel<<<GRID1(E * BTH * 16), THREADS, 0, stream>>>(
            src, dst, featb, evals, ssum, bufA, E, BTH, tH, tdim);
        bias_relu_kernel<<<GRID1(aggN / 4), THREADS, 0, stream>>>((float4*)bufA, b2, aggN / 4);
        // LN2 over (N,C) per (b, 16) : bufA -> bufB  [B,N,16,C]
        ln_nc_kernel<<<BB * 16, THREADS, 0, stream>>>(bufA, bufB, ln2w, ln2b, 16);
    }

    // ===================== Output layer =====================
    {
        const int KT = 5;                 // (H-1)*T + 1
        const int Mc = BB * NN * 12;      // 196608
        // tconv1 as 5 accumulated GEMMs: bufB [B,N,16,C] -> bufA [B,N,12,C]
        for (int k = 0; k < KT; ++k) {
            wmma_gemm_kernel<<<dim3(Mc / 16, CC / 32), 32, 0, stream>>>(
                bufB, tc1w, (k == KT - 1) ? tc1b : nullptr, bufA,
                Mc, CC, CC,
                /*rpg_out*/12, /*rpg_in*/16, /*row_off*/k,
                /*b_sk(ci)*/KT, /*b_sn(co)*/CC * KT, /*b_off*/k,
                /*acc*/(k > 0), /*relu*/0);
        }
        // LN3 over (N,C) per (b, 12) : bufA -> bufB  [B,N,12,C]
        ln_nc_kernel<<<BB * 12, THREADS, 0, stream>>>(bufA, bufB, ln3w, ln3b, 12);
        // tconv2 (1x1): bufB -> bufA  [B,N,12,C];  B'[ci,co] = tc2w[co*64+ci]
        wmma_gemm_kernel<<<dim3(Mc / 16, CC / 32), 32, 0, stream>>>(
            bufB, tc2w, tc2b, bufA, Mc, CC, CC, 1, 1, 0, 1, CC, 0, 0, 0);
        // torch-faithful reshape into [B*N, 768]
        const int totalR = BB * NN * 768;
        repack_fc_kernel<<<GRID1(totalR), THREADS, 0, stream>>>(bufA, featb, totalR);
        // fc: [B*N,768] @ fc_w^T + fc_b -> d_out [B,N,12]
        const int Mf = BB * NN;
        wmma_gemm_kernel<<<dim3(Mf / 16, 1), 32, 0, stream>>>(
            featb, fcw, fcb, (float*)d_out, Mf, 768, 12, 1, 1, 0, 1, 768, 0, 0, 0);
    }
    #undef GRID1
}
